// diff_attention_15917148799505
// MI455X (gfx1250) — matrix operand order repaired and checked by reading, not compiled, not run on hardware
//
#include <hip/hip_runtime.h>
#include <hip/hip_bf16.h>

typedef __bf16 bf16;
typedef __attribute__((ext_vector_type(16))) __bf16 bf16x16;
typedef __attribute__((ext_vector_type(8)))  __bf16 bf16x8;
typedef __attribute__((ext_vector_type(8)))  float  f32x8;

#define NROWS 8192
#define DDIM  1024

// ---------- WMMA fragment helpers (gfx1250 layouts, 16x16x32 bf16) ----------

// A matrix 16x32 (MxK), row-major in memory with leading dim ld.
// lane r(0..15), half h: elements 0..7 = K[8h .. 8h+7], 8..15 = K[8h+16 .. 8h+23]
__device__ __forceinline__ bf16x16 load_frag_a(const bf16* __restrict__ base, int ld) {
  const int lane = threadIdx.x & 31;
  const int r = lane & 15, h = lane >> 4;
  const bf16* p = base + (long)r * ld + h * 8;
  bf16x8 lo = *reinterpret_cast<const bf16x8*>(p);
  bf16x8 hi = *reinterpret_cast<const bf16x8*>(p + 16);
  bf16x16 f;
#pragma unroll
  for (int i = 0; i < 8; ++i) { f[i] = lo[i]; f[i + 8] = hi[i]; }
  return f;
}

// B matrix 32x16 (KxN) where element(k, col) = M[col*ld + k] (K-contiguous per column).
// lane r(0..15) = column, half h: elements 0..7 = K[8h .. 8h+7], 8..15 = K[8h+16 .. 8h+23]
__device__ __forceinline__ bf16x16 load_frag_b(const bf16* __restrict__ base, long ld) {
  const int lane = threadIdx.x & 31;
  const int r = lane & 15, h = lane >> 4;
  const bf16* p = base + (long)r * ld + h * 8;
  bf16x8 lo = *reinterpret_cast<const bf16x8*>(p);
  bf16x8 hi = *reinterpret_cast<const bf16x8*>(p + 16);
  bf16x16 f;
#pragma unroll
  for (int i = 0; i < 8; ++i) { f[i] = lo[i]; f[i + 8] = hi[i]; }
  return f;
}

__device__ __forceinline__ f32x8 wmma_bf16(bf16x16 a, bf16x16 b, f32x8 c) {
  return __builtin_amdgcn_wmma_f32_16x16x32_bf16(false, a, false, b, (short)0, c, false, false);
}

// Online softmax row-stat update for one 16x16 score tile (C layout: elem i of
// lane (r,h) = S[i+8h][r]; rows live across the 16 lanes of each half).
__device__ __forceinline__ void update_stats(const f32x8& s, float scale,
                                             float* __restrict__ M, float* __restrict__ L) {
#pragma unroll
  for (int i = 0; i < 8; ++i) {
    float v = s[i] * scale;
    float rmax = v;
    rmax = fmaxf(rmax, __shfl_xor(rmax, 1, 32));
    rmax = fmaxf(rmax, __shfl_xor(rmax, 2, 32));
    rmax = fmaxf(rmax, __shfl_xor(rmax, 4, 32));
    rmax = fmaxf(rmax, __shfl_xor(rmax, 8, 32));
    float nm = fmaxf(M[i], rmax);
    float e = __expf(v - nm);
    float rs = e;
    rs += __shfl_xor(rs, 1, 32);
    rs += __shfl_xor(rs, 2, 32);
    rs += __shfl_xor(rs, 4, 32);
    rs += __shfl_xor(rs, 8, 32);
    L[i] = L[i] * __expf(M[i] - nm) + rs;
    M[i] = nm;
  }
}

// ---------- fp32 -> bf16 conversion / packing ----------
__global__ __launch_bounds__(256) void cvt_f32_bf16_strided(
    const float* __restrict__ src, bf16* __restrict__ dst,
    int rows, int cols, int ldDst) {
  long idx = (long)blockIdx.x * blockDim.x + threadIdx.x;
  long total = (long)rows * cols;
  if (idx >= total) return;
  long r = idx / cols, c = idx - r * cols;
  dst[r * (long)ldDst + c] = (bf16)src[idx];
}

// ---------- GEMM: C[m,n] = sum_k A[m,k]*W[n,k] + bias[n]  (y = x @ W.T + b) ----------
// block = 256 threads (8 waves, 4x2), block tile 128x128, each wave 32x64.
// 8 WMMAs per k-step on 6 fragment loads (A-frags reused over 4 B-frags).
__global__ __launch_bounds__(256) void gemm_bias_bf16(
    const bf16* __restrict__ A, int lda,
    const bf16* __restrict__ W, int ldw,
    const float* __restrict__ bias,
    bf16* __restrict__ C, int ldc, int K) {
  const int wave = threadIdx.x >> 5;
  const int lane = threadIdx.x & 31;
  const int r = lane & 15, h = lane >> 4;
  const int m0 = blockIdx.x * 128 + (wave & 3) * 32;
  const int n0 = blockIdx.y * 128 + (wave >> 2) * 64;
  f32x8 acc[2][4];
#pragma unroll
  for (int im = 0; im < 2; ++im)
#pragma unroll
    for (int in = 0; in < 4; ++in) acc[im][in] = f32x8{};

  for (int k = 0; k < K; k += 32) {
    bf16x16 a0 = load_frag_a(A + (long)m0 * lda + k, lda);
    bf16x16 a1 = load_frag_a(A + (long)(m0 + 16) * lda + k, lda);
#pragma unroll
    for (int in = 0; in < 4; ++in) {
      bf16x16 b = load_frag_b(W + (long)(n0 + 16 * in) * ldw + k, ldw);
      acc[0][in] = wmma_bf16(a0, b, acc[0][in]);
      acc[1][in] = wmma_bf16(a1, b, acc[1][in]);
    }
  }
#pragma unroll
  for (int im = 0; im < 2; ++im) {
#pragma unroll
    for (int in = 0; in < 4; ++in) {
#pragma unroll
      for (int i = 0; i < 8; ++i) {
        int row = m0 + 16 * im + i + 8 * h;
        int col = n0 + 16 * in + r;
        C[(long)row * ldc + col] = (bf16)(acc[im][in][i] + bias[col]);
      }
    }
  }
}

// ---------- bf16 transpose (V[N,D] -> Vt[D,N]) via LDS ----------
__global__ __launch_bounds__(256) void transpose_bf16_k(
    const bf16* __restrict__ in, bf16* __restrict__ out, int rows, int cols) {
  __shared__ bf16 tile[32][33];
  const int tx = threadIdx.x, ty = threadIdx.y;
  const int c0 = blockIdx.x * 32, r0 = blockIdx.y * 32;
#pragma unroll
  for (int j = 0; j < 32; j += 8)
    tile[ty + j][tx] = in[(long)(r0 + ty + j) * cols + c0 + tx];
  __syncthreads();
#pragma unroll
  for (int j = 0; j < 32; j += 8)
    out[(long)(c0 + ty + j) * rows + r0 + tx] = tile[tx][ty + j];
}

// ---------- Pass 1: softmax row stats (max, denom) for both attention streams ----------
// 8 waves/block, each wave owns 16 query rows, streams keys in 32-wide tiles so each
// Q A-fragment feeds two K B-fragments (4 WMMAs per 6 fragment loads).
__global__ __launch_bounds__(256) void attn_stats_kernel(
    const bf16* __restrict__ Q1, const bf16* __restrict__ K1,
    const bf16* __restrict__ Q2, const bf16* __restrict__ K2,
    float* __restrict__ m1, float* __restrict__ l1,
    float* __restrict__ m2, float* __restrict__ l2) {
  const int wave = threadIdx.x >> 5;
  const int lane = threadIdx.x & 31;
  const int r = lane & 15, h = lane >> 4;
  const int q0 = (blockIdx.x * 8 + wave) * 16;
  const float scale = 0.03125f; // 1/sqrt(1024)
  float M1[8], L1[8], M2[8], L2[8];
#pragma unroll
  for (int i = 0; i < 8; ++i) { M1[i] = -3.0e38f; L1[i] = 0.f; M2[i] = -3.0e38f; L2[i] = 0.f; }

  for (int j0 = 0; j0 < NROWS; j0 += 32) {
    // prefetch next key tile while we compute on this one
    if (j0 + 32 < NROWS) {
      __builtin_prefetch(K1 + (long)(j0 + 32) * DDIM + lane * 64, 0, 3);
      __builtin_prefetch(K2 + (long)(j0 + 32) * DDIM + lane * 64, 0, 3);
    }
    f32x8 s1a{}, s1b{}, s2a{}, s2b{};
    for (int d = 0; d < DDIM; d += 32) {
      bf16x16 a1 = load_frag_a(Q1 + (long)q0 * DDIM + d, DDIM);
      bf16x16 b1a = load_frag_b(K1 + (long)j0 * DDIM + d, DDIM);
      bf16x16 b1b = load_frag_b(K1 + (long)(j0 + 16) * DDIM + d, DDIM);
      s1a = wmma_bf16(a1, b1a, s1a);
      s1b = wmma_bf16(a1, b1b, s1b);
      bf16x16 a2 = load_frag_a(Q2 + (long)q0 * DDIM + d, DDIM);
      bf16x16 b2a = load_frag_b(K2 + (long)j0 * DDIM + d, DDIM);
      bf16x16 b2b = load_frag_b(K2 + (long)(j0 + 16) * DDIM + d, DDIM);
      s2a = wmma_bf16(a2, b2a, s2a);
      s2b = wmma_bf16(a2, b2b, s2b);
    }
    update_stats(s1a, scale, M1, L1);
    update_stats(s1b, scale, M1, L1);
    update_stats(s2a, scale, M2, L2);
    update_stats(s2b, scale, M2, L2);
  }
  if (r == 0) {
#pragma unroll
    for (int i = 0; i < 8; ++i) {
      int row = q0 + i + 8 * h;
      m1[row] = M1[i]; l1[row] = L1[i];
      m2[row] = M2[i]; l2[row] = L2[i];
    }
  }
}

// ---------- Pass 2: out = (softmax1 - alpha*softmax2) @ V, fused via LDS P tile ----------
// 512 threads (16 waves). Block owns 16 query rows. Per 256-key block:
//   each wave computes S1,S2 for its 16-key subtile, writes combined P (bf16) to LDS,
//   then every wave does P(16x256) @ Vt-slice(256x64) WMMAs for its private 64 output cols.
__global__ __launch_bounds__(512) void attn_apply_kernel(
    const bf16* __restrict__ Q1, const bf16* __restrict__ K1,
    const bf16* __restrict__ Q2, const bf16* __restrict__ K2,
    const bf16* __restrict__ Vt,
    const float* __restrict__ m1, const float* __restrict__ l1,
    const float* __restrict__ m2, const float* __restrict__ l2,
    const float* __restrict__ alphaPtr, float* __restrict__ out) {
  __shared__ bf16 Pl[16 * 256];
  const int wave = threadIdx.x >> 5;
  const int lane = threadIdx.x & 31;
  const int r = lane & 15, h = lane >> 4;
  const int q0 = blockIdx.x * 16;
  const int d0 = wave * 64;
  const float scale = 0.03125f;
  const float alpha = alphaPtr[0];

  float rm1[8], ri1[8], rm2[8], ri2[8];
#pragma unroll
  for (int i = 0; i < 8; ++i) {
    int row = q0 + i + 8 * h;
    rm1[i] = m1[row]; ri1[i] = 1.0f / l1[row];
    rm2[i] = m2[row]; ri2[i] = 1.0f / l2[row];
  }

  f32x8 o0{}, o1{}, o2{}, o3{};
  for (int kb = 0; kb < NROWS; kb += 256) {
    const int j0 = kb + wave * 16;
    // prefetch next key block's K rows for this wave
    if (kb + 256 < NROWS) {
      __builtin_prefetch(K1 + (long)(j0 + 256) * DDIM + lane * 64, 0, 3);
      __builtin_prefetch(K2 + (long)(j0 + 256) * DDIM + lane * 64, 0, 3);
    }
    f32x8 s1{}, s2{};
    for (int d = 0; d < DDIM; d += 32) {
      bf16x16 a1 = load_frag_a(Q1 + (long)q0 * DDIM + d, DDIM);
      bf16x16 b1 = load_frag_b(K1 + (long)j0 * DDIM + d, DDIM);
      s1 = wmma_bf16(a1, b1, s1);
      bf16x16 a2 = load_frag_a(Q2 + (long)q0 * DDIM + d, DDIM);
      bf16x16 b2 = load_frag_b(K2 + (long)j0 * DDIM + d, DDIM);
      s2 = wmma_bf16(a2, b2, s2);
    }
#pragma unroll
    for (int i = 0; i < 8; ++i) {
      float e1 = __expf(s1[i] * scale - rm1[i]) * ri1[i];
      float e2 = __expf(s2[i] * scale - rm2[i]) * ri2[i];
      Pl[(i + 8 * h) * 256 + wave * 16 + r] = (bf16)(e1 - alpha * e2);
    }
    __syncthreads();
#pragma unroll
    for (int kt = 0; kt < 256; kt += 32) {
      bf16x16 a = load_frag_a(&Pl[kt], 256);
      bf16x16 vb0 = load_frag_b(Vt + (long)(d0 +  0) * NROWS + kb + kt, NROWS);
      bf16x16 vb1 = load_frag_b(Vt + (long)(d0 + 16) * NROWS + kb + kt, NROWS);
      bf16x16 vb2 = load_frag_b(Vt + (long)(d0 + 32) * NROWS + kb + kt, NROWS);
      bf16x16 vb3 = load_frag_b(Vt + (long)(d0 + 48) * NROWS + kb + kt, NROWS);
      o0 = wmma_bf16(a, vb0, o0);
      o1 = wmma_bf16(a, vb1, o1);
      o2 = wmma_bf16(a, vb2, o2);
      o3 = wmma_bf16(a, vb3, o3);
    }
    __syncthreads();
  }
#pragma unroll
  for (int i = 0; i < 8; ++i) {
    int row = q0 + i + 8 * h;
    out[(long)row * DDIM + d0 +  0 + r] = o0[i];
    out[(long)row * DDIM + d0 + 16 + r] = o1[i];
    out[(long)row * DDIM + d0 + 32 + r] = o2[i];
    out[(long)row * DDIM + d0 + 48 + r] = o3[i];
  }
}

// ---------------------------- host launcher ----------------------------
extern "C" void kernel_launch(void* const* d_in, const int* in_sizes, int n_in,
                              void* d_out, int out_size, void* d_ws, size_t ws_size,
                              hipStream_t stream) {
  const float* X1 = (const float*)d_in[0];
  const float* X2 = (const float*)d_in[1];
  const float* Wq = (const float*)d_in[2];
  const float* bq = (const float*)d_in[3];
  const float* Wk = (const float*)d_in[4];
  const float* bk = (const float*)d_in[5];
  const float* Wv = (const float*)d_in[6];
  const float* bv = (const float*)d_in[7];
  const float* alpha = (const float*)d_in[8];
  float* out = (float*)d_out;

  char* ws = (char*)d_ws;
  size_t off = 0;
  auto wsalloc = [&](size_t bytes) -> void* {
    void* p = ws + off;
    off += (bytes + 255) & ~(size_t)255;
    return p;
  };
  bf16* Xb  = (bf16*)wsalloc((size_t)NROWS * 2048 * 2); // X1 || X2, bf16
  bf16* Wqb = (bf16*)wsalloc((size_t)1024 * 1024 * 2);
  bf16* Wkb = (bf16*)wsalloc((size_t)1024 * 1024 * 2);
  bf16* Wvb = (bf16*)wsalloc((size_t)1024 * 2048 * 2);
  bf16* Q1  = (bf16*)wsalloc((size_t)NROWS * DDIM * 2);
  bf16* K1  = (bf16*)wsalloc((size_t)NROWS * DDIM * 2);
  bf16* Q2  = (bf16*)wsalloc((size_t)NROWS * DDIM * 2);
  bf16* K2  = (bf16*)wsalloc((size_t)NROWS * DDIM * 2);
  bf16* Vb  = (bf16*)wsalloc((size_t)NROWS * DDIM * 2);
  bf16* Vt  = (bf16*)wsalloc((size_t)DDIM * NROWS * 2);
  float* m1 = (float*)wsalloc((size_t)NROWS * 4);
  float* l1 = (float*)wsalloc((size_t)NROWS * 4);
  float* m2 = (float*)wsalloc((size_t)NROWS * 4);
  float* l2 = (float*)wsalloc((size_t)NROWS * 4);

  // 1) bf16 conversions / packing
  {
    long nx = (long)NROWS * DDIM;
    int gx = (int)((nx + 255) / 256);
    cvt_f32_bf16_strided<<<gx, 256, 0, stream>>>(X1, Xb, NROWS, DDIM, 2048);
    cvt_f32_bf16_strided<<<gx, 256, 0, stream>>>(X2, Xb + DDIM, NROWS, DDIM, 2048);
    long nw = (long)1024 * 1024;
    int gw = (int)((nw + 255) / 256);
    cvt_f32_bf16_strided<<<gw, 256, 0, stream>>>(Wq, Wqb, 1024, 1024, 1024);
    cvt_f32_bf16_strided<<<gw, 256, 0, stream>>>(Wk, Wkb, 1024, 1024, 1024);
    long nv = (long)1024 * 2048;
    int gv = (int)((nv + 255) / 256);
    cvt_f32_bf16_strided<<<gv, 256, 0, stream>>>(Wv, Wvb, 1024, 2048, 2048);
  }

  // 2) projections (WMMA GEMMs), block tile 128x128
  dim3 gg(NROWS / 128, DDIM / 128);
  gemm_bias_bf16<<<gg, 256, 0, stream>>>(Xb,        2048, Wqb, 1024, bq, Q1, DDIM, 1024);
  gemm_bias_bf16<<<gg, 256, 0, stream>>>(Xb,        2048, Wkb, 1024, bk, K1, DDIM, 1024);
  gemm_bias_bf16<<<gg, 256, 0, stream>>>(Xb + DDIM, 2048, Wqb, 1024, bq, Q2, DDIM, 1024);
  gemm_bias_bf16<<<gg, 256, 0, stream>>>(Xb + DDIM, 2048, Wkb, 1024, bk, K2, DDIM, 1024);
  gemm_bias_bf16<<<gg, 256, 0, stream>>>(Xb,        2048, Wvb, 2048, bv, Vb, DDIM, 2048);

  // 3) V transpose for contiguous PV B-fragments
  transpose_bf16_k<<<dim3(DDIM / 32, NROWS / 32), dim3(32, 8), 0, stream>>>(Vb, Vt, NROWS, DDIM);

  // 4) softmax statistics (flash pass 1, both streams)
  attn_stats_kernel<<<NROWS / 128, 256, 0, stream>>>(Q1, K1, Q2, K2, m1, l1, m2, l2);

  // 5) fused (A1 - alpha*A2) @ V (flash pass 2)
  attn_apply_kernel<<<NROWS / 16, 512, 0, stream>>>(Q1, K1, Q2, K2, Vt,
                                                    m1, l1, m2, l2, alpha, out);
}
